// RoiHeads_36558761624222
// MI455X (gfx1250) — compile-verified, matrix operand order checked by reading, not run on hardware
//
#include <hip/hip_runtime.h>
#include <hip/hip_bf16.h>

// ---------------- problem constants (match reference) ----------------
#define BB        8
#define PP        8192
#define CC        91
#define CM1       90                 // classes minus background
#define MM        (PP * CM1)         // 737280 candidates per image
#define ROWS      (BB * PP)          // 65536 proposal rows
#define TOPK      1000
#define DETS      100
#define CAP       4096               // collection cap per image (power of 2)
#define NBIN      1024
#define NEGV      (-1.0e9f)
#define SCORE_TH  0.05f
#define MIN_SZ    0.01f
#define XCLIP     4.135166556742356f // log(1000/16)

typedef __attribute__((ext_vector_type(16))) _Float16 v16h;
typedef __attribute__((ext_vector_type(8)))  float    v8f;

// ---------------- ws layout (bytes) ----------------
#define OFF_HIST   0                        // 8*1024 u32      = 32768
#define OFF_CNT    (64 * 1024)              // 8 int
#define OFF_T      (OFF_CNT + 4096)         // 8 int
#define OFF_DENOM  (128 * 1024)             // 65536 f32       = 256KB
#define OFF_SKEY   (OFF_DENOM + ROWS * 4)   // 8*737280 f32    = 23.6MB
#define OFF_KEYS   (OFF_SKEY + (size_t)BB * MM * 4)      // 8*4096 u64 = 256KB
#define OFF_SELIDX (OFF_KEYS + (size_t)BB * CAP * 8)     // 8*1000 int (pad 4096)
#define OFF_SELSC  (OFF_SELIDX + 32768)
#define OFF_BOXD   (OFF_SELSC + 32768)      // 8*1000 float4
#define OFF_BOXO   (OFF_BOXD + 131072)      // 8*1000 float4 (class-offset)
#define OFF_LABEL  (OFF_BOXO + 131072)      // 8*1000 int
#define OFF_VALID  (OFF_LABEL + 32768)      // 8*1000 int

// =====================================================================
// K0: zero histogram + counters
// =====================================================================
__global__ void k0_init(unsigned int* hist, int* cnt) {
  int i = blockIdx.x * blockDim.x + threadIdx.x;
  if (i < BB * NBIN) hist[i] = 0u;
  if (i < BB)        cnt[i]  = 0;
}

// =====================================================================
// K1: softmax denominators via WMMA ones-matmul reduction.
// One wave32 handles 16 proposal rows; 3x v_wmma_f32_16x16x32_f16
// accumulate row-sums of exp(logit) over classes 0..95 (91..95 padded 0).
// All gathers are unconditional (addresses clamped, pads zeroed by select)
// so loads batch without EXEC branching and a single wait.
// =====================================================================
__global__ __launch_bounds__(32) void k1_denom_wmma(
    const float* __restrict__ logits, float* __restrict__ denom) {
  const int lane  = threadIdx.x;
  const int g     = lane >> 4;            // half-wave group per A-layout
  const int rbase = blockIdx.x * 16;
  const int row   = rbase + (lane & 15);
  const float* lp = logits + (long)row * CC;

  v16h ones;
#pragma unroll
  for (int e = 0; e < 16; ++e) ones[e] = (_Float16)1.0f;

  // ---- phase 1: unconditional gathers into registers ----
  // element e<8  -> class k = 32t + 8g + e      (8 consecutive)
  // element e>=8 -> class k = 32t + 16 + 8g + e-8 (8 consecutive)
  float raw[3][16];
#pragma unroll
  for (int t = 0; t < 3; ++t) {
    const int b0 = 32 * t + 8 * g;
#pragma unroll
    for (int e = 0; e < 8; ++e) {
      int k0 = b0 + e;
      int k1 = b0 + 16 + e;
      if (t == 2) {                      // only chunk 2 can exceed C-1
        k0 = k0 < CC ? k0 : (CC - 1);
        k1 = k1 < CC ? k1 : (CC - 1);
      }
      raw[t][e]     = lp[k0];
      raw[t][8 + e] = lp[k1];
    }
  }

  // ---- phase 2: exp, pad-zero via value select, 3 accumulating WMMAs ----
  v8f acc = {};
#pragma unroll
  for (int t = 0; t < 3; ++t) {
    v16h a;
#pragma unroll
    for (int e = 0; e < 16; ++e) {
      float v = __expf(raw[t][e]);
      if (t == 2) {
        int k = 64 + ((e < 8) ? (8 * g + e) : (16 + 8 * g + (e - 8)));
        a[e] = (k < CC) ? (_Float16)v : (_Float16)0.0f;  // cndmask, no branch
      } else {
        a[e] = (_Float16)v;
      }
    }
    acc = __builtin_amdgcn_wmma_f32_16x16x32_f16(
        false, a, false, ones, (short)0, acc, false, false);
  }

  // stage next tile toward the caches (global_prefetch_b8)
  __builtin_prefetch(logits + (long)(rbase + 16) * CC, 0, 1);

  // D layout: lane0 VGPR j = row j (j<8); lane16 VGPR j = row 8+j
  if (lane == 0) {
#pragma unroll
    for (int j = 0; j < 8; ++j) denom[rbase + j] = acc[j];
  } else if (lane == 16) {
#pragma unroll
    for (int j = 0; j < 8; ++j) denom[rbase + 8 + j] = acc[j];
  }
}

// =====================================================================
// shared decode helper
// =====================================================================
__device__ __forceinline__ float4 decode_box(float4 pr, float4 rel,
                                             float Wf, float Hf) {
  float w  = pr.z - pr.x, h = pr.w - pr.y;
  float cx = pr.x + 0.5f * w, cy = pr.y + 0.5f * h;
  float dx = rel.x * 0.1f, dy = rel.y * 0.1f;
  float dw = fminf(rel.z * 0.2f, XCLIP);
  float dh = fminf(rel.w * 0.2f, XCLIP);
  float pcx = dx * w + cx, pcy = dy * h + cy;
  float pw  = __expf(dw) * w, ph = __expf(dh) * h;
  float4 o;
  o.x = fminf(fmaxf(pcx - 0.5f * pw, 0.f), Wf);
  o.y = fminf(fmaxf(pcy - 0.5f * ph, 0.f), Hf);
  o.z = fminf(fmaxf(pcx + 0.5f * pw, 0.f), Wf);
  o.w = fminf(fmaxf(pcy + 0.5f * ph, 0.f), Hf);
  return o;
}

// =====================================================================
// K2: per-candidate score + validity + score key + histogram
// =====================================================================
__global__ void k2_score(const float* __restrict__ logits,
                         const float* __restrict__ reg,
                         const float* __restrict__ props,
                         const float* __restrict__ denom,
                         const int* __restrict__ hw,
                         float* __restrict__ skey,
                         unsigned int* __restrict__ hist) {
  int i = blockIdx.x * blockDim.x + threadIdx.x;
  if (i >= BB * MM) return;
  int b = i / MM;
  int m = i - b * MM;
  int p = m / CM1;
  int c = 1 + (m - p * CM1);
  int row = b * PP + p;

  float Hf = (float)hw[0], Wf = (float)hw[1];
  float sc = __expf(logits[(long)row * CC + c]) / denom[row];
  float4 rel = ((const float4*)(reg + (long)row * (4 * CC)))[c];
  float4 pr  = ((const float4*)props)[row];
  float4 bx  = decode_box(pr, rel, Wf, Hf);

  bool valid = (sc > SCORE_TH) && ((bx.z - bx.x) >= MIN_SZ) &&
               ((bx.w - bx.y) >= MIN_SZ);
  skey[i] = valid ? sc : NEGV;
  if (valid) {
    int bin = (int)(sc * (float)NBIN);
    bin = bin > (NBIN - 1) ? (NBIN - 1) : bin;
    atomicAdd(&hist[b * NBIN + bin], 1u);
  }
}

// =====================================================================
// K3: per-image cutoff bin for top-1000
// =====================================================================
__global__ void k3_thresh(const unsigned int* __restrict__ hist,
                          int* __restrict__ T) {
  int b = threadIdx.x;
  if (b >= BB) return;
  const unsigned int* h = hist + b * NBIN;
  int cum = 0, t = 0;
  for (int bin = NBIN - 1; bin >= 0; --bin) {
    cum += (int)h[bin];
    if (cum >= TOPK) { t = bin; break; }
  }
  T[b] = t;
}

// =====================================================================
// K4: collect survivors as 64-bit (score, ~idx) keys (deterministic sort key)
// =====================================================================
__global__ void k4_collect(const float* __restrict__ skey,
                           const int* __restrict__ T,
                           int* __restrict__ cnt,
                           unsigned long long* __restrict__ keys) {
  int i = blockIdx.x * blockDim.x + threadIdx.x;
  if (i >= BB * MM) return;
  float s = skey[i];
  if (s <= 0.0f) return;
  int b = i / MM;
  int bin = (int)(s * (float)NBIN);
  bin = bin > (NBIN - 1) ? (NBIN - 1) : bin;
  if (bin < T[b]) return;
  int pos = atomicAdd(&cnt[b], 1);
  if (pos < CAP) {
    unsigned int m  = (unsigned int)(i - b * MM);
    unsigned int fe = __float_as_uint(s) | 0x80000000u; // s>0: order-preserving
    keys[(long)b * CAP + pos] =
        ((unsigned long long)fe << 32) | (unsigned int)(~m);
  }
}

// =====================================================================
// K5: per-image bitonic sort (desc) of 4096 keys in LDS -> top-1000
// =====================================================================
__global__ __launch_bounds__(1024) void k5_sort(
    const unsigned long long* __restrict__ keys,
    const int* __restrict__ cnt,
    int* __restrict__ selidx, float* __restrict__ selsc) {
  __shared__ unsigned long long sk[CAP];
  int b = blockIdx.x, tid = threadIdx.x;
  int n = cnt[b]; n = n > CAP ? CAP : n;
#pragma unroll
  for (int w = 0; w < CAP / 1024; ++w) {
    int t = tid + w * 1024;
    sk[t] = (t < n) ? keys[(long)b * CAP + t] : 0ULL;
  }
  for (unsigned k = 2; k <= CAP; k <<= 1) {
    for (unsigned j = k >> 1; j > 0; j >>= 1) {
      __syncthreads();
#pragma unroll
      for (int w = 0; w < CAP / 1024; ++w) {
        unsigned i = tid + w * 1024;
        unsigned ixj = i ^ j;
        if (ixj > i) {
          unsigned long long a = sk[i], c = sk[ixj];
          bool up = ((i & k) == 0);               // descending overall
          if (up ? (a < c) : (a > c)) { sk[i] = c; sk[ixj] = a; }
        }
      }
    }
  }
  __syncthreads();
  if (tid < TOPK) {
    unsigned long long key = sk[tid];
    int gi = b * TOPK + tid;
    if (key != 0ULL) {
      unsigned int m = ~((unsigned int)(key & 0xFFFFFFFFu));
      unsigned int u = (unsigned int)(key >> 32) & 0x7FFFFFFFu;
      selidx[gi] = (int)m;
      selsc[gi]  = __uint_as_float(u);
    } else {
      selidx[gi] = -1;
      selsc[gi]  = NEGV;
    }
  }
}

// =====================================================================
// K6: decode boxes only for the 8x1000 survivors (+ class-offset boxes)
// =====================================================================
__global__ __launch_bounds__(1024) void k6_decode_sel(
    const float* __restrict__ reg, const float* __restrict__ props,
    const int* __restrict__ hw, const int* __restrict__ selidx,
    float4* __restrict__ boxd, float4* __restrict__ boxo,
    int* __restrict__ label, int* __restrict__ vflag) {
  int t = threadIdx.x;
  if (t >= TOPK) return;
  int b = blockIdx.x, gi = b * TOPK + t;
  int idx = selidx[gi];
  if (idx < 0) {
    boxd[gi] = make_float4(0.f, 0.f, 0.f, 0.f);
    boxo[gi] = make_float4(0.f, 0.f, 0.f, 0.f);
    label[gi] = 0; vflag[gi] = 0;
    return;
  }
  float Hf = (float)hw[0], Wf = (float)hw[1];
  float os = fmaxf(Hf, Wf) + 1.0f;
  int p = idx / CM1;
  int c = 1 + (idx - p * CM1);
  int row = b * PP + p;
  float4 rel = ((const float4*)(reg + (long)row * (4 * CC)))[c];
  float4 pr  = ((const float4*)props)[row];
  float4 bx  = decode_box(pr, rel, Wf, Hf);
  boxd[gi] = bx;
  float off = (float)c * os;
  boxo[gi] = make_float4(bx.x + off, bx.y + off, bx.z + off, bx.w + off);
  label[gi] = c; vflag[gi] = 1;
}

// =====================================================================
// K7: greedy NMS (reference order) + ordered compaction to 100 outputs
// =====================================================================
__global__ __launch_bounds__(1024) void k7_nms_out(
    const float4* __restrict__ boxd, const float4* __restrict__ boxo,
    const float* __restrict__ selsc, const int* __restrict__ label,
    const int* __restrict__ vflag, float* __restrict__ out) {
  __shared__ float4 sbo[TOPK];
  __shared__ int ssup[TOPK];
  __shared__ int sval[TOPK];
  __shared__ int rank[TOPK];
  int b = blockIdx.x, tid = threadIdx.x;
  for (int t = tid; t < TOPK; t += 1024) {
    sbo[t]  = boxo[b * TOPK + t];
    ssup[t] = 0;
    sval[t] = vflag[b * TOPK + t];
  }
  __syncthreads();
  for (int i = 0; i < TOPK; ++i) {
    bool act = sval[i] && !ssup[i];
    if (act && tid > i && tid < TOPK) {
      float4 bi = sbo[i], bj = sbo[tid];
      float xx1 = fmaxf(bi.x, bj.x), yy1 = fmaxf(bi.y, bj.y);
      float xx2 = fminf(bi.z, bj.z), yy2 = fminf(bi.w, bj.w);
      float iw = fmaxf(xx2 - xx1, 0.f), ih = fmaxf(yy2 - yy1, 0.f);
      float inter = iw * ih;
      float ai = (bi.z - bi.x) * (bi.w - bi.y);
      float aj = (bj.z - bj.x) * (bj.w - bj.y);
      float iou = inter / (ai + aj - inter + 1e-9f);
      if (iou > 0.5f) ssup[tid] = 1;
    }
    __syncthreads();
  }
  if (tid == 0) {                       // ordered compaction ranks
    int r = 0;
    for (int t = 0; t < TOPK; ++t)
      rank[t] = (sval[t] && !ssup[t] && r < DETS) ? r++ : -1;
  }
  if (tid < DETS) {                     // zero defaults
    int base = (b * DETS + tid) * 4;
    out[base + 0] = 0.f; out[base + 1] = 0.f;
    out[base + 2] = 0.f; out[base + 3] = 0.f;
    out[BB * DETS * 4 + b * DETS + tid] = 0.f;            // score
    out[BB * DETS * 5 + b * DETS + tid] = 0.f;            // label
  }
  __syncthreads();
  if (tid < TOPK && rank[tid] >= 0) {
    int r = rank[tid], gi = b * TOPK + tid;
    float4 bx = boxd[gi];
    int base = (b * DETS + r) * 4;
    out[base + 0] = bx.x; out[base + 1] = bx.y;
    out[base + 2] = bx.z; out[base + 3] = bx.w;
    out[BB * DETS * 4 + b * DETS + r] = selsc[gi];
    out[BB * DETS * 5 + b * DETS + r] = (float)label[gi];
  }
}

// =====================================================================
extern "C" void kernel_launch(void* const* d_in, const int* in_sizes, int n_in,
                              void* d_out, int out_size, void* d_ws,
                              size_t ws_size, hipStream_t stream) {
  const float* logits = (const float*)d_in[0];
  const float* reg    = (const float*)d_in[1];
  const float* props  = (const float*)d_in[2];
  const int*   hw     = (const int*)d_in[3];
  float* out = (float*)d_out;
  char* ws = (char*)d_ws;

  unsigned int* hist = (unsigned int*)(ws + OFF_HIST);
  int*   cnt    = (int*)(ws + OFF_CNT);
  int*   T      = (int*)(ws + OFF_T);
  float* denom  = (float*)(ws + OFF_DENOM);
  float* skey   = (float*)(ws + OFF_SKEY);
  unsigned long long* keys = (unsigned long long*)(ws + OFF_KEYS);
  int*   selidx = (int*)(ws + OFF_SELIDX);
  float* selsc  = (float*)(ws + OFF_SELSC);
  float4* boxd  = (float4*)(ws + OFF_BOXD);
  float4* boxo  = (float4*)(ws + OFF_BOXO);
  int*   label  = (int*)(ws + OFF_LABEL);
  int*   vflag  = (int*)(ws + OFF_VALID);

  k0_init<<<(BB * NBIN + 255) / 256, 256, 0, stream>>>(hist, cnt);
  k1_denom_wmma<<<ROWS / 16, 32, 0, stream>>>(logits, denom);

  int ncand = BB * MM;
  int blk2 = (ncand + 255) / 256;
  k2_score<<<blk2, 256, 0, stream>>>(logits, reg, props, denom, hw, skey, hist);
  k3_thresh<<<1, 32, 0, stream>>>(hist, T);
  k4_collect<<<blk2, 256, 0, stream>>>(skey, T, cnt, keys);
  k5_sort<<<BB, 1024, 0, stream>>>(keys, cnt, selidx, selsc);
  k6_decode_sel<<<BB, 1024, 0, stream>>>(reg, props, hw, selidx, boxd, boxo,
                                         label, vflag);
  k7_nms_out<<<BB, 1024, 0, stream>>>(boxd, boxo, selsc, label, vflag, out);
}
